// HATT_16587163697552
// MI455X (gfx1250) — matrix operand order repaired and checked by reading, not compiled, not run on hardware
//
#include <hip/hip_runtime.h>
#include <hip/hip_bf16.h>

typedef __attribute__((ext_vector_type(16))) _Float16 v16h;
typedef __attribute__((ext_vector_type(8)))  float    v8f;
typedef __attribute__((ext_vector_type(4)))  float    v4f;

#define H_HEADS 8
#define DK_DIM 32
#define NDIM 256
#define INV_SQRT_DK 0.17677669529663687f

// ---- order-preserving float <-> uint mapping for atomicMax-based segment max ----
__device__ __forceinline__ unsigned f2ord(float f) {
  unsigned u = __float_as_uint(f);
  return (u & 0x80000000u) ? ~u : (u | 0x80000000u);   // any real float encodes > 0
}
__device__ __forceinline__ float ord2f(unsigned e) {
  unsigned u = (e & 0x80000000u) ? (e ^ 0x80000000u) : ~e;
  return __uint_as_float(u);
}

// ---------------------------------------------------------------------------
// Prep: fuse rel_att into Wk (block-diagonal right-multiply) and its bias,
// transpose all weights to (N,K) layout, split into f16 hi + f16 lo.
// ---------------------------------------------------------------------------
__global__ void hgt_prep(const float* __restrict__ Wk, const float* __restrict__ bk,
                         const float* __restrict__ Wq, const float* __restrict__ bq,
                         const float* __restrict__ ratt,   // (2, 8, 32, 32)
                         _Float16* __restrict__ WtKhi, _Float16* __restrict__ WtKlo,
                         _Float16* __restrict__ WtQhi, _Float16* __restrict__ WtQlo,
                         float* __restrict__ bK, float* __restrict__ bQ) {
  int t = blockIdx.x * blockDim.x + threadIdx.x;
  if (t >= 2 * NDIM * NDIM) return;
  int rrel = t >> 16;            // relation
  int n    = (t >> 8) & 255;     // output column (head*32 + e)
  int k    = t & 255;            // input dim
  int st = rrel, dt = 1 - rrel;  // rel0: src=A(type0),dst=B(type1); rel1: swapped
  int head = n >> 5, e = n & 31;

  // W'[k][h*32+e] = sum_d Wk[st][k][h*32+d] * ratt[rrel][h][d][e]
  const float* wkrow = Wk + ((size_t)st * NDIM + k) * NDIM + head * 32;
  const float* ra    = ratt + (((size_t)rrel * H_HEADS + head) * 32) * 32 + e;
  float acc = 0.f;
#pragma unroll 8
  for (int d = 0; d < 32; ++d) acc += wkrow[d] * ra[d * 32];

  size_t o = ((size_t)rrel * NDIM + n) * NDIM + k;   // (rel, N, K) transposed
  _Float16 hv = (_Float16)acc;
  WtKhi[o] = hv;
  WtKlo[o] = (_Float16)(acc - (float)hv);

  float qv = Wq[((size_t)dt * NDIM + k) * NDIM + n];
  _Float16 qh = (_Float16)qv;
  WtQhi[o] = qh;
  WtQlo[o] = (_Float16)(qv - (float)qh);

  if (k == 0) {   // fused bias: b'[n] = sum_d bk[st][h*32+d]*ratt[h][d][e]
    const float* bkrow = bk + (size_t)st * NDIM + head * 32;
    float bacc = 0.f;
#pragma unroll 8
    for (int d = 0; d < 32; ++d) bacc += bkrow[d] * ra[d * 32];
    bK[rrel * NDIM + n] = bacc;
    bQ[rrel * NDIM + n] = bq[(size_t)dt * NDIM + n];
  }
}

__global__ void hgt_zero(unsigned* __restrict__ p, int n) {
  int t = blockIdx.x * blockDim.x + threadIdx.x;
  if (t < n) p[t] = 0u;
}

// ---------------------------------------------------------------------------
// Projection GEMM: Out(M x 256) = Hsrc(M x 256, f32) @ W(256 x 256) + bias
// W stored transposed (N,K) as f16 hi/lo pair. One wave -> 16x128 output tile
// (8 f32 accumulators), split-f16 product: ah*bh + ah*bl + al*bh (err ~2^-21).
//
// A-frag layout (16-bit 16x32): lane r=lane&15 is row M; c=lane>>4 selects
//   contiguous K chunk [c*16, c*16+16).
// B-frag layout (32x16): lane r is column N, contiguous K chunk [c*16, c*16+16).
// C/D layout: VGPR j, lanes 0-15 -> row j, lanes 16-31 -> row j+8; col = lane&15.
//
// All loads use hoisted per-lane base pointers + compile-time immediate
// offsets (K loop fully unrolled) so no per-iteration 64-bit address VALU.
// ---------------------------------------------------------------------------
__global__ __launch_bounds__(256)
void hgt_proj_gemm(const float* __restrict__ Hsrc, int M,
                   const _Float16* __restrict__ Whi,
                   const _Float16* __restrict__ Wlo,
                   const float* __restrict__ bias,
                   float* __restrict__ Out) {
  const int lane = threadIdx.x & 31;
  const int wave = threadIdx.x >> 5;
  const int m0 = (blockIdx.x * 8 + wave) * 16;
  if (m0 >= M) return;                       // wave-uniform exit, EXEC stays all-1 for WMMA
  const int n0 = blockIdx.y * 128;
  const int r = lane & 15;
  const int c = lane >> 4;

  const int mr = m0 + r;
  const float*    aptr  = Hsrc + (size_t)(mr < M ? mr : 0) * NDIM + c * 16;
  const _Float16* bhptr = Whi + (size_t)(n0 + r) * NDIM + c * 16;
  const _Float16* blptr = Wlo + (size_t)(n0 + r) * NDIM + c * 16;

  v8f acc[8] = {};

#pragma unroll
  for (int kb = 0; kb < NDIM; kb += 32) {
    // A fragment: f32 -> split f16 hi + lo. Contiguous K = kb+c*16 .. kb+c*16+15.
    v4f x0 = *(const v4f*)(aptr + kb);
    v4f x1 = *(const v4f*)(aptr + kb + 4);
    v4f x2 = *(const v4f*)(aptr + kb + 8);
    v4f x3 = *(const v4f*)(aptr + kb + 12);
    v16h ahi, alo;
#pragma unroll
    for (int j = 0; j < 4; ++j) {
      _Float16 h0 = (_Float16)x0[j], h1 = (_Float16)x1[j];
      _Float16 h2 = (_Float16)x2[j], h3 = (_Float16)x3[j];
      ahi[j]      = h0;  alo[j]      = (_Float16)(x0[j] - (float)h0);
      ahi[4 + j]  = h1;  alo[4 + j]  = (_Float16)(x1[j] - (float)h1);
      ahi[8 + j]  = h2;  alo[8 + j]  = (_Float16)(x2[j] - (float)h2);
      ahi[12 + j] = h3;  alo[12 + j] = (_Float16)(x3[j] - (float)h3);
    }
#pragma unroll
    for (int t = 0; t < 8; ++t) {
      // constant element offset: t*16 rows of W (t*16*NDIM) + kb
      v16h bhi = *(const v16h*)(bhptr + t * 16 * NDIM + kb);   // 32B contiguous
      v16h blo = *(const v16h*)(blptr + t * 16 * NDIM + kb);
      acc[t] = __builtin_amdgcn_wmma_f32_16x16x32_f16(false, ahi, false, bhi,
                                                      (short)0, acc[t], false, false);
      acc[t] = __builtin_amdgcn_wmma_f32_16x16x32_f16(false, ahi, false, blo,
                                                      (short)0, acc[t], false, false);
      acc[t] = __builtin_amdgcn_wmma_f32_16x16x32_f16(false, alo, false, bhi,
                                                      (short)0, acc[t], false, false);
    }
  }

#pragma unroll
  for (int t = 0; t < 8; ++t) {
    const int col = n0 + t * 16 + r;
    const float bv = bias[col];
#pragma unroll
    for (int j = 0; j < 8; ++j) {
      const int row = m0 + j + 8 * c;
      if (row < M) Out[(size_t)row * NDIM + col] = acc[t][j] + bv;
    }
  }
}

// ---------------------------------------------------------------------------
// Per-edge score: t[e,h] = (q[dst[e],h,:] . k[src[e],h,:]) * pri[h] / sqrt(DK)
// Also accumulates segment max via order-preserving uint atomicMax.
// ---------------------------------------------------------------------------
__global__ void hgt_edge_score(const float* __restrict__ Q, const float* __restrict__ K,
                               const int* __restrict__ src, const int* __restrict__ dst,
                               const float* __restrict__ pri, int E,
                               float* __restrict__ score, unsigned* __restrict__ maxb) {
  int t = blockIdx.x * blockDim.x + threadIdx.x;
  if (t >= E * H_HEADS) return;
  int e = t >> 3, h = t & 7;
  int d = dst[e], s = src[e];
  const float* qp = Q + (size_t)d * NDIM + h * DK_DIM;
  const float* kp = K + (size_t)s * NDIM + h * DK_DIM;
  float acc = 0.f;
#pragma unroll
  for (int j = 0; j < DK_DIM; j += 4) {
    v4f qv = *(const v4f*)(qp + j);
    v4f kv = *(const v4f*)(kp + j);
    acc += qv[0] * kv[0] + qv[1] * kv[1] + qv[2] * kv[2] + qv[3] * kv[3];
  }
  acc *= pri[h] * INV_SQRT_DK;
  score[t] = acc;
  atomicMax(&maxb[(size_t)d * H_HEADS + h], f2ord(acc));
}

__global__ void hgt_edge_exp(const int* __restrict__ dst, int E,
                             float* __restrict__ score,
                             const unsigned* __restrict__ maxb,
                             float* __restrict__ sumb) {
  int t = blockIdx.x * blockDim.x + threadIdx.x;
  if (t >= E * H_HEADS) return;
  int e = t >> 3, h = t & 7;
  int d = dst[e];
  float m = ord2f(maxb[(size_t)d * H_HEADS + h]);
  float z = expf(score[t] - m);
  score[t] = z;
  atomicAdd(&sumb[(size_t)d * H_HEADS + h], z);
}

__global__ void hgt_edge_norm(const int* __restrict__ dst, int E,
                              float* __restrict__ score,
                              const float* __restrict__ sumb) {
  int t = blockIdx.x * blockDim.x + threadIdx.x;
  if (t >= E * H_HEADS) return;
  int e = t >> 3, h = t & 7;
  score[t] = score[t] / sumb[(size_t)dst[e] * H_HEADS + h];
}

// ---------------------------------------------------------------------------
extern "C" void kernel_launch(void* const* d_in, const int* in_sizes, int n_in,
                              void* d_out, int out_size, void* d_ws, size_t ws_size,
                              hipStream_t stream) {
  const float* h_A  = (const float*)d_in[0];
  const float* h_B  = (const float*)d_in[1];
  const float* Wk   = (const float*)d_in[2];
  const float* bk   = (const float*)d_in[3];
  const float* Wq   = (const float*)d_in[4];
  const float* bq   = (const float*)d_in[5];
  const float* pri  = (const float*)d_in[6];   // (2, 8)
  const float* ratt = (const float*)d_in[7];   // (2, 8, 32, 32)
  const int*   src0 = (const int*)d_in[8];
  const int*   dst0 = (const int*)d_in[9];
  const int*   src1 = (const int*)d_in[10];
  const int*   dst1 = (const int*)d_in[11];

  const int N_A = in_sizes[0] / NDIM;
  const int N_B = in_sizes[1] / NDIM;
  const int E0  = in_sizes[8];
  const int E1  = in_sizes[10];
  float* out = (float*)d_out;

  // ---- workspace layout (256B-aligned chunks) ----
  char* w = (char*)d_ws;
  auto take = [&](size_t bytes) -> char* {
    char* p = w;
    w += (bytes + 255) & ~(size_t)255;
    return p;
  };
  _Float16* WtKhi = (_Float16*)take((size_t)2 * NDIM * NDIM * sizeof(_Float16));
  _Float16* WtKlo = (_Float16*)take((size_t)2 * NDIM * NDIM * sizeof(_Float16));
  _Float16* WtQhi = (_Float16*)take((size_t)2 * NDIM * NDIM * sizeof(_Float16));
  _Float16* WtQlo = (_Float16*)take((size_t)2 * NDIM * NDIM * sizeof(_Float16));
  float* bK = (float*)take((size_t)2 * NDIM * sizeof(float));
  float* bQ = (float*)take((size_t)2 * NDIM * sizeof(float));
  float* K0 = (float*)take((size_t)N_A * NDIM * sizeof(float)); // proj(h_A, Wk0*ratt0)
  float* Q0 = (float*)take((size_t)N_B * NDIM * sizeof(float)); // proj(h_B, Wq1)
  float* K1 = (float*)take((size_t)N_B * NDIM * sizeof(float)); // proj(h_B, Wk1*ratt1)
  float* Q1 = (float*)take((size_t)N_A * NDIM * sizeof(float)); // proj(h_A, Wq0)
  unsigned* maxb = (unsigned*)take((size_t)(N_A + N_B) * H_HEADS * sizeof(unsigned));
  float*    sumb = (float*)   take((size_t)(N_A + N_B) * H_HEADS * sizeof(float));

  dim3 blk(256);

  // 1) weight prep (fuse rel_att into K-path, transpose, f16 hi/lo split)
  hgt_prep<<<dim3((2 * NDIM * NDIM + 255) / 256), blk, 0, stream>>>(
      Wk, bk, Wq, bq, ratt, WtKhi, WtKlo, WtQhi, WtQlo, bK, bQ);

  // 2) zero segment-reduction buffers (max encodes as 0 == -infinity sentinel)
  {
    int nz = (N_A + N_B) * H_HEADS * 2;   // maxb + sumb are contiguous dwords
    hgt_zero<<<dim3((nz + 255) / 256), blk, 0, stream>>>(maxb, nz);
  }

  // 3) four projection GEMMs (WMMA f16 hi/lo, f32 accumulate), 16x128 per wave
  dim3 gA((((N_A + 15) / 16) + 7) / 8, NDIM / 128);
  dim3 gB((((N_B + 15) / 16) + 7) / 8, NDIM / 128);
  hgt_proj_gemm<<<gA, blk, 0, stream>>>(h_A, N_A, WtKhi,               WtKlo,               bK,        K0);
  hgt_proj_gemm<<<gB, blk, 0, stream>>>(h_B, N_B, WtQhi,               WtQlo,               bQ,        Q0);
  hgt_proj_gemm<<<gB, blk, 0, stream>>>(h_B, N_B, WtKhi + NDIM * NDIM, WtKlo + NDIM * NDIM, bK + NDIM, K1);
  hgt_proj_gemm<<<gA, blk, 0, stream>>>(h_A, N_A, WtQhi + NDIM * NDIM, WtQlo + NDIM * NDIM, bQ + NDIM, Q1);

  // 4) relation 0: A -> B (dst nodes are B)
  {
    dim3 ge((E0 * H_HEADS + 255) / 256);
    float*    s0 = out;
    unsigned* m0 = maxb;
    float*    z0 = sumb;
    hgt_edge_score<<<ge, blk, 0, stream>>>(Q0, K0, src0, dst0, pri, E0, s0, m0);
    hgt_edge_exp  <<<ge, blk, 0, stream>>>(dst0, E0, s0, m0, z0);
    hgt_edge_norm <<<ge, blk, 0, stream>>>(dst0, E0, s0, z0);
  }
  // 5) relation 1: B -> A (dst nodes are A)
  {
    dim3 ge((E1 * H_HEADS + 255) / 256);
    float*    s1 = out + (size_t)E0 * H_HEADS;
    unsigned* m1 = maxb + (size_t)N_B * H_HEADS;
    float*    z1 = sumb + (size_t)N_B * H_HEADS;
    hgt_edge_score<<<ge, blk, 0, stream>>>(Q1, K1, src1, dst1, pri + H_HEADS, E1, s1, m1);
    hgt_edge_exp  <<<ge, blk, 0, stream>>>(dst1, E1, s1, m1, z1);
    hgt_edge_norm <<<ge, blk, 0, stream>>>(dst1, E1, s1, z1);
  }
}
